// TransformerBlock_22136261444057
// MI455X (gfx1250) — matrix operand order repaired and checked by reading, not compiled, not run on hardware
//
#include <hip/hip_runtime.h>
#include <hip/hip_bf16.h>
#include <math.h>

// ---------------------------------------------------------------------------
// Types for CDNA5 WMMA (wave32): v_wmma_f32_16x16x32_bf16
// ---------------------------------------------------------------------------
typedef __bf16 bf16_t;
typedef __attribute__((ext_vector_type(16))) bf16_t v16bf;
typedef __attribute__((ext_vector_type(8)))  bf16_t bf16x8;   // 16-byte chunk
typedef __attribute__((ext_vector_type(8)))  float  v8f;

#define N_EMB  1024
#define BATCH  4
#define SEQ    2048
#define HEADS  16
#define HDIM   64
#define ROWS   (BATCH * SEQ)      // 8192
#define LN_EPS 1e-3f

__device__ __forceinline__ bf16_t f2bf(float f) {
  unsigned u = __builtin_bit_cast(unsigned, f);
  unsigned r = u + 0x7FFFu + ((u >> 16) & 1u);     // RNE
  unsigned short h = (unsigned short)(r >> 16);
  return __builtin_bit_cast(bf16_t, h);
}

// LDS segment byte-offset of a generic pointer into a __shared__ array
__device__ __forceinline__ unsigned lds_off(const void* p) {
  return (unsigned)(unsigned long long)(__attribute__((address_space(3))) const void*)p;
}

// CDNA5 async DMA: global -> LDS, 16 bytes, tracked on ASYNCcnt
__device__ __forceinline__ void async_copy_b128(const void* gptr, const void* lptr) {
  asm volatile("global_load_async_to_lds_b128 %0, %1, off"
               :: "v"(lds_off(lptr)), "v"((unsigned long long)gptr)
               : "memory");
}
__device__ __forceinline__ void wait_asynccnt0() {
  asm volatile("s_wait_asynccnt 0x0" ::: "memory");
}

// ---------------------------------------------------------------------------
// fp32 -> bf16 weight conversion (grid-stride)
// ---------------------------------------------------------------------------
__global__ void cvt_f32_bf16_kernel(const float* __restrict__ src,
                                    bf16_t* __restrict__ dst, int n) {
  int i = blockIdx.x * blockDim.x + threadIdx.x;
  int stride = gridDim.x * blockDim.x;
  for (; i < n; i += stride) dst[i] = f2bf(src[i]);
}

// ---------------------------------------------------------------------------
// LayerNorm: one block (256 thr = 8 waves) per row of 1024, bf16 output
// ---------------------------------------------------------------------------
__global__ __launch_bounds__(256)
void layernorm_bf16_kernel(const float* __restrict__ x,
                           const float* __restrict__ gamma,
                           const float* __restrict__ beta,
                           bf16_t* __restrict__ out) {
  __shared__ float red_s[2][8];
  const int row = blockIdx.x;
  const int t = threadIdx.x;
  const float* xr = x + (size_t)row * N_EMB;

  float v[4], s = 0.f, ss = 0.f;
#pragma unroll
  for (int i = 0; i < 4; ++i) {
    v[i] = xr[t + i * 256];
    s += v[i];
    ss += v[i] * v[i];
  }
#pragma unroll
  for (int m = 1; m < 32; m <<= 1) {
    s  += __shfl_xor(s,  m, 32);
    ss += __shfl_xor(ss, m, 32);
  }
  if ((t & 31) == 0) { red_s[0][t >> 5] = s; red_s[1][t >> 5] = ss; }
  __syncthreads();
  float S = 0.f, SS = 0.f;
#pragma unroll
  for (int i = 0; i < 8; ++i) { S += red_s[0][i]; SS += red_s[1][i]; }
  const float mean = S * (1.0f / N_EMB);
  const float var  = SS * (1.0f / N_EMB) - mean * mean;
  const float inv  = rsqrtf(var + LN_EPS);

  bf16_t* orow = out + (size_t)row * N_EMB;
#pragma unroll
  for (int i = 0; i < 4; ++i) {
    int c = t + i * 256;
    orow[c] = f2bf(gamma[c] * (v[i] - mean) * inv + beta[c]);
  }
}

// ---------------------------------------------------------------------------
// BF16 WMMA GEMM: out[M,N] = A[M,K] @ B[K,N] (+bias)(GELU)(+resid)
// Block tile 128x128, BK=32, 256 threads = 8 waves (4x2), wave tile 32x64.
// A tile staged via async global->LDS DMA (ASYNCcnt); B tile staged through
// VGPRs with an in-register transpose into N-major LDS. Double buffered,
// one barrier per K-step. Rows padded to 40 bf16 (80B) vs 64 banks.
// ---------------------------------------------------------------------------
template <bool BIAS, bool RESID, bool GELU, bool OUTBF>
__global__ __launch_bounds__(256)
void gemm_bf16_kernel(const bf16_t* __restrict__ A, const bf16_t* __restrict__ B,
                      const float* __restrict__ bias, const float* __restrict__ resid,
                      float* __restrict__ outF, bf16_t* __restrict__ outB,
                      int M, int N, int K) {
  constexpr int AP = 40;  // padded row pitch (bf16)
  __shared__ bf16_t a_s[2][128 * AP];
  __shared__ bf16_t b_s[2][128 * AP];

  const int tid  = threadIdx.x;
  const int lane = tid & 31;
  const int wv   = tid >> 5;
  const int wm   = wv & 3;          // 4 waves down  -> rows
  const int wn   = wv >> 2;         // 2 waves across-> cols
  const int m16  = lane & 15;       // M/N index within 16x16 fragment
  const int hi   = lane >> 4;       // K-half selector (contiguous k = 16*hi + s)
  const int bm   = blockIdx.y * 128;
  const int bn   = blockIdx.x * 128;

  v8f acc[2][4];
#pragma unroll
  for (int mi = 0; mi < 2; ++mi)
#pragma unroll
    for (int ni = 0; ni < 4; ++ni)
#pragma unroll
      for (int r = 0; r < 8; ++r) acc[mi][ni][r] = 0.f;

  bf16x8 breg[2];
  int ar_[2], ac_[2], bk_[2], bn_[2];
#pragma unroll
  for (int c = 0; c < 2; ++c) {
    int lin = tid + c * 256;                      // 0..511
    ar_[c] = lin >> 2; ac_[c] = (lin & 3) << 3;   // A: 128 rows x 4 chunks
    bk_[c] = lin >> 4; bn_[c] = (lin & 15) << 3;  // B: 32 k-rows x 16 chunks
  }

  auto asyncA = [&](int kt, int st) {             // async DMA, contiguous copy
#pragma unroll
    for (int c = 0; c < 2; ++c)
      async_copy_b128(A + (size_t)(bm + ar_[c]) * K + kt * 32 + ac_[c],
                      &a_s[st][ar_[c] * AP + ac_[c]]);
  };
  auto gloadB = [&](int kt) {
#pragma unroll
    for (int c = 0; c < 2; ++c)
      breg[c] = *(const bf16x8*)(B + (size_t)(kt * 32 + bk_[c]) * N + bn + bn_[c]);
  };
  auto sstoreB = [&](int st) {
#pragma unroll
    for (int c = 0; c < 2; ++c)
#pragma unroll
      for (int j = 0; j < 8; ++j)                 // transpose B into N-major
        b_s[st][(bn_[c] + j) * AP + bk_[c]] = breg[c][j];
  };
  auto compute = [&](int st) {
    v16bf af[2], bfr[4];
#pragma unroll
    for (int mi = 0; mi < 2; ++mi) {
      const bf16_t* p = &a_s[st][(wm * 32 + mi * 16 + m16) * AP];
      bf16x8 lo = *(const bf16x8*)(p + hi * 16);
      bf16x8 h8 = *(const bf16x8*)(p + hi * 16 + 8);
#pragma unroll
      for (int j = 0; j < 8; ++j) { af[mi][j] = lo[j]; af[mi][8 + j] = h8[j]; }
    }
#pragma unroll
    for (int ni = 0; ni < 4; ++ni) {
      const bf16_t* p = &b_s[st][(wn * 64 + ni * 16 + m16) * AP + hi * 16];
      bf16x8 lo = *(const bf16x8*)(p);
      bf16x8 h8 = *(const bf16x8*)(p + 8);
#pragma unroll
      for (int j = 0; j < 8; ++j) { bfr[ni][j] = lo[j]; bfr[ni][8 + j] = h8[j]; }
    }
#pragma unroll
    for (int mi = 0; mi < 2; ++mi)
#pragma unroll
      for (int ni = 0; ni < 4; ++ni)
        acc[mi][ni] = __builtin_amdgcn_wmma_f32_16x16x32_bf16(
            false, af[mi], false, bfr[ni], (short)0, acc[mi][ni], false, false);
  };

  const int KT = K >> 5;
  asyncA(0, 0);
  gloadB(0);
  sstoreB(0);
  wait_asynccnt0();
  __syncthreads();
  for (int kt = 1; kt < KT; ++kt) {
    // stage kt&1 was last READ by compute() at iteration kt-1, which is
    // separated from these writes by the end-of-iteration barrier.
    asyncA(kt, kt & 1);
    gloadB(kt);
    compute((kt - 1) & 1);
    sstoreB(kt & 1);
    wait_asynccnt0();          // own async writes done before signaling
    __syncthreads();
  }
  compute((KT - 1) & 1);

  // epilogue
#pragma unroll
  for (int mi = 0; mi < 2; ++mi)
#pragma unroll
    for (int ni = 0; ni < 4; ++ni) {
      const int c = bn + wn * 64 + ni * 16 + m16;
      const float bv = BIAS ? bias[c] : 0.f;
#pragma unroll
      for (int r = 0; r < 8; ++r) {
        const int row = bm + wm * 32 + mi * 16 + r + hi * 8;
        float v = acc[mi][ni][r] + bv;
        if (GELU)  v = 0.5f * v * (1.0f + erff(v * 0.70710678118654752f));
        if (RESID) v += resid[(size_t)row * N + c];
        if (OUTBF) outB[(size_t)row * N + c] = f2bf(v);
        else       outF[(size_t)row * N + c] = v;
      }
    }
}

// ---------------------------------------------------------------------------
// Causal flash attention over packed qkv bf16 [B*T, 3C].
// Grid (T/64, H, B); 128 thr = 4 waves; wave owns 16 q-rows x full D=64.
// K tile staged via async DMA (it is already the WMMA B-operand layout);
// V tile transposed through VGPRs.
// ---------------------------------------------------------------------------
__global__ __launch_bounds__(128)
void flash_attn_kernel(const bf16_t* __restrict__ qkv, bf16_t* __restrict__ attn) {
  constexpr int KP = 72;                    // padded pitch (bf16)
  __shared__ bf16_t k_s[64 * KP];           // K tile, row-major [t][d]
  __shared__ bf16_t vT_s[64 * KP];          // V tile transposed [d][t]
  __shared__ bf16_t p_s[4][16 * KP];        // per-wave P scratch

  const int tid  = threadIdx.x;
  const int lane = tid & 31;
  const int wv   = tid >> 5;
  const int m16  = lane & 15;
  const int hi   = lane >> 4;

  const int qblk = blockIdx.x, h = blockIdx.y, b = blockIdx.z;
  const int q0   = qblk * 64 + wv * 16;
  const size_t rowBase = (size_t)b * SEQ;
  const int C3 = 3 * N_EMB;
  const float scale = 0.125f;               // 1/sqrt(64)

  // Q fragments (A layout), straight from global
  v16bf qf[2];
#pragma unroll
  for (int ki = 0; ki < 2; ++ki) {
    const bf16_t* qp = qkv + (rowBase + q0 + m16) * C3 + h * HDIM + ki * 32;
    bf16x8 lo = *(const bf16x8*)(qp + hi * 16);
    bf16x8 h8 = *(const bf16x8*)(qp + hi * 16 + 8);
#pragma unroll
    for (int j = 0; j < 8; ++j) { qf[ki][j] = lo[j]; qf[ki][8 + j] = h8[j]; }
  }

  v8f o[4];
  float mrow[8], lrow[8];
#pragma unroll
  for (int dj = 0; dj < 4; ++dj)
#pragma unroll
    for (int r = 0; r < 8; ++r) o[dj][r] = 0.f;
#pragma unroll
  for (int r = 0; r < 8; ++r) { mrow[r] = -3.0e38f; lrow[r] = 0.f; }

  const int nkb = qblk + 1;                 // causal block bound (uniform)
  for (int kb = 0; kb < nkb; ++kb) {
    const int k0 = kb * 64;
    // cooperative K/V tile load: K via async DMA, V transposed via VGPRs
#pragma unroll
    for (int c = 0; c < 4; ++c) {
      int lin = tid + c * 128;              // 0..511
      int r = lin >> 3, cc = (lin & 7) << 3;
      const bf16_t* base = qkv + (rowBase + k0 + r) * C3 + h * HDIM;
      async_copy_b128(base + N_EMB + cc, &k_s[r * KP + cc]);
      bf16x8 vv = *(const bf16x8*)(base + 2 * N_EMB + cc);
#pragma unroll
      for (int j = 0; j < 8; ++j) vT_s[(cc + j) * KP + r] = vv[j];
    }
    wait_asynccnt0();
    __syncthreads();

    // S = Q @ K^T   (K tile row-major IS the B operand layout)
    v8f S[4];
#pragma unroll
    for (int nj = 0; nj < 4; ++nj)
#pragma unroll
      for (int r = 0; r < 8; ++r) S[nj][r] = 0.f;
#pragma unroll
    for (int ki = 0; ki < 2; ++ki)
#pragma unroll
      for (int nj = 0; nj < 4; ++nj) {
        const bf16_t* p = &k_s[(nj * 16 + m16) * KP + ki * 32 + hi * 16];
        bf16x8 lo = *(const bf16x8*)(p);
        bf16x8 h8 = *(const bf16x8*)(p + 8);
        v16bf bb;
#pragma unroll
        for (int j = 0; j < 8; ++j) { bb[j] = lo[j]; bb[8 + j] = h8[j]; }
        S[nj] = __builtin_amdgcn_wmma_f32_16x16x32_bf16(
            false, qf[ki], false, bb, (short)0, S[nj], false, false);
      }

    // scale + causal mask
#pragma unroll
    for (int nj = 0; nj < 4; ++nj) {
      const int t = k0 + nj * 16 + m16;
#pragma unroll
      for (int r = 0; r < 8; ++r) {
        const int q = q0 + r + hi * 8;
        S[nj][r] = (t <= q) ? S[nj][r] * scale : -1.0e30f;
      }
    }

    // online softmax (row stats: 16-lane reductions within each half)
    float bmax[8];
#pragma unroll
    for (int r = 0; r < 8; ++r) {
      float mx = S[0][r];
#pragma unroll
      for (int nj = 1; nj < 4; ++nj) mx = fmaxf(mx, S[nj][r]);
      bmax[r] = mx;
    }
#pragma unroll
    for (int m = 1; m < 16; m <<= 1)
#pragma unroll
      for (int r = 0; r < 8; ++r) bmax[r] = fmaxf(bmax[r], __shfl_xor(bmax[r], m, 32));

    float alpha[8];
#pragma unroll
    for (int r = 0; r < 8; ++r) {
      float mnew = fmaxf(mrow[r], bmax[r]);
      alpha[r] = expf(mrow[r] - mnew);
      mrow[r] = mnew;
      lrow[r] *= alpha[r];
    }
#pragma unroll
    for (int dj = 0; dj < 4; ++dj)
#pragma unroll
      for (int r = 0; r < 8; ++r) o[dj][r] *= alpha[r];

    // P = exp(S - m), row sums, spill to LDS for A-layout reload
    float psum[8];
#pragma unroll
    for (int r = 0; r < 8; ++r) psum[r] = 0.f;
#pragma unroll
    for (int nj = 0; nj < 4; ++nj)
#pragma unroll
      for (int r = 0; r < 8; ++r) {
        float p = expf(S[nj][r] - mrow[r]);
        psum[r] += p;
        p_s[wv][(r + hi * 8) * KP + nj * 16 + m16] = f2bf(p);
      }
#pragma unroll
    for (int m = 1; m < 16; m <<= 1)
#pragma unroll
      for (int r = 0; r < 8; ++r) psum[r] += __shfl_xor(psum[r], m, 32);
#pragma unroll
    for (int r = 0; r < 8; ++r) lrow[r] += psum[r];

    // O += P @ V  (V transposed in LDS -> contiguous B fragments)
#pragma unroll
    for (int kk = 0; kk < 2; ++kk) {
      const bf16_t* pp = &p_s[wv][m16 * KP + kk * 32];
      bf16x8 lo = *(const bf16x8*)(pp + hi * 16);
      bf16x8 h8 = *(const bf16x8*)(pp + hi * 16 + 8);
      v16bf pf;
#pragma unroll
      for (int j = 0; j < 8; ++j) { pf[j] = lo[j]; pf[8 + j] = h8[j]; }
#pragma unroll
      for (int dj = 0; dj < 4; ++dj) {
        const bf16_t* vp = &vT_s[(dj * 16 + m16) * KP + kk * 32 + hi * 16];
        bf16x8 vlo = *(const bf16x8*)(vp);
        bf16x8 vh8 = *(const bf16x8*)(vp + 8);
        v16bf vb;
#pragma unroll
        for (int j = 0; j < 8; ++j) { vb[j] = vlo[j]; vb[8 + j] = vh8[j]; }
        o[dj] = __builtin_amdgcn_wmma_f32_16x16x32_bf16(
            false, pf, false, vb, (short)0, o[dj], false, false);
      }
    }
    __syncthreads();
  }

  // finalize: O / l -> bf16 attn output [B*T, C]
#pragma unroll
  for (int dj = 0; dj < 4; ++dj)
#pragma unroll
    for (int r = 0; r < 8; ++r) {
      const size_t row = rowBase + q0 + r + hi * 8;
      attn[row * N_EMB + h * HDIM + dj * 16 + m16] = f2bf(o[dj][r] / lrow[r]);
    }
}

// ---------------------------------------------------------------------------
// Host-side launcher
// ---------------------------------------------------------------------------
extern "C" void kernel_launch(void* const* d_in, const int* in_sizes, int n_in,
                              void* d_out, int out_size, void* d_ws, size_t ws_size,
                              hipStream_t stream) {
  const float* x      = (const float*)d_in[0];
  const float* w_qkv  = (const float*)d_in[1];
  const float* w_proj = (const float*)d_in[2];
  const float* b_proj = (const float*)d_in[3];
  const float* w1     = (const float*)d_in[4];
  const float* b1     = (const float*)d_in[5];
  const float* w2     = (const float*)d_in[6];
  const float* b2     = (const float*)d_in[7];
  const float* gamma1 = (const float*)d_in[8];
  const float* beta1  = (const float*)d_in[9];
  const float* gamma2 = (const float*)d_in[10];
  const float* beta2  = (const float*)d_in[11];
  float* out = (float*)d_out;

  char* ws = (char*)d_ws;
  size_t off = 0;
  auto walloc = [&](size_t bytes) -> void* {
    void* p = ws + off;
    off += (bytes + 255) & ~(size_t)255;
    return p;
  };
  bf16_t* wqkv_b = (bf16_t*)walloc((size_t)N_EMB * 3 * N_EMB * 2);
  bf16_t* wprj_b = (bf16_t*)walloc((size_t)N_EMB * N_EMB * 2);
  bf16_t* w1_b   = (bf16_t*)walloc((size_t)N_EMB * 4 * N_EMB * 2);
  bf16_t* w2_b   = (bf16_t*)walloc((size_t)4 * N_EMB * N_EMB * 2);
  bf16_t* hA_b   = (bf16_t*)walloc((size_t)ROWS * N_EMB * 2);
  bf16_t* qkv_b  = (bf16_t*)walloc((size_t)ROWS * 3 * N_EMB * 2);
  bf16_t* attn_b = (bf16_t*)walloc((size_t)ROWS * N_EMB * 2);
  float*  x2_f   = (float*)walloc((size_t)ROWS * N_EMB * 4);
  bf16_t* hB_b   = (bf16_t*)walloc((size_t)ROWS * N_EMB * 2);
  bf16_t* g_b    = (bf16_t*)walloc((size_t)ROWS * 4 * N_EMB * 2);

  auto cvt = [&](const float* src, bf16_t* dst, int n) {
    int blocks = (n + 1023) / 1024;
    cvt_f32_bf16_kernel<<<blocks, 256, 0, stream>>>(src, dst, n);
  };
  cvt(w_qkv,  wqkv_b, N_EMB * 3 * N_EMB);
  cvt(w_proj, wprj_b, N_EMB * N_EMB);
  cvt(w1,     w1_b,   N_EMB * 4 * N_EMB);
  cvt(w2,     w2_b,   4 * N_EMB * N_EMB);

  // LN1
  layernorm_bf16_kernel<<<ROWS, 256, 0, stream>>>(x, gamma1, beta1, hA_b);

  // QKV: [8192,1024] @ [1024,3072] -> bf16
  gemm_bf16_kernel<false, false, false, true>
      <<<dim3(3 * N_EMB / 128, ROWS / 128), 256, 0, stream>>>(
          hA_b, wqkv_b, nullptr, nullptr, nullptr, qkv_b, ROWS, 3 * N_EMB, N_EMB);

  // causal flash attention
  flash_attn_kernel<<<dim3(SEQ / 64, HEADS, BATCH), 128, 0, stream>>>(qkv_b, attn_b);

  // proj + b_proj + residual(x) -> x2 (f32)
  gemm_bf16_kernel<true, true, false, false>
      <<<dim3(N_EMB / 128, ROWS / 128), 256, 0, stream>>>(
          attn_b, wprj_b, b_proj, x, x2_f, nullptr, ROWS, N_EMB, N_EMB);

  // LN2
  layernorm_bf16_kernel<<<ROWS, 256, 0, stream>>>(x2_f, gamma2, beta2, hB_b);

  // FFN1 + b1 + exact GELU -> bf16
  gemm_bf16_kernel<true, false, true, true>
      <<<dim3(4 * N_EMB / 128, ROWS / 128), 256, 0, stream>>>(
          hB_b, w1_b, b1, nullptr, nullptr, g_b, ROWS, 4 * N_EMB, N_EMB);

  // FFN2 + b2 + residual(x2) -> d_out (f32)
  gemm_bf16_kernel<true, true, false, false>
      <<<dim3(N_EMB / 128, ROWS / 128), 256, 0, stream>>>(
          g_b, w2_b, b2, x2_f, out, nullptr, ROWS, N_EMB, 4 * N_EMB);
}
